// Net_61838939128428
// MI455X (gfx1250) — matrix operand order repaired and checked by reading, not compiled, not run on hardware
//
#include <hip/hip_runtime.h>
#include <hip/hip_bf16.h>
#include <math.h>

// ---------------------------------------------------------------------------
// CDNA5 (gfx1250) wave32 WMMA: D = A(16x32 f16) x B(32x16 f16) + C(f32)
// All LDS tiles are kept in *fragment order*: one 32x16-half block per k-tile
// (per (kt,nt) for weights), so a fragment load is 32 contiguous bytes/lane
// -> two ds_load_b128, no per-element packing.
// Activations (e, x) are stored in HBM as f16: they are only ever consumed as
// f16 WMMA inputs, and this cuts hot-loop traffic ~40% on a BW-bound kernel.
// ---------------------------------------------------------------------------
typedef __attribute__((ext_vector_type(16))) _Float16 v16h;
typedef __attribute__((ext_vector_type(8)))  float    v8f;

#define WAVES_PER_BLOCK 4   // 128 threads/block, 16 rows per wave
#define FRAG 512            // halves per fragment block (32 lanes x 16 halves)

__device__ __forceinline__ v8f wmma_f32_f16(v16h a, v16h b, v8f c) {
  // (neg_a, A, neg_b, B, c_mod, C, reuse_a, reuse_b)
  return __builtin_amdgcn_wmma_f32_16x16x32_f16(false, a, false, b, (short)0, c,
                                                false, false);
}

// Load one fragment (A or B) that is stored in fragment order in LDS.
__device__ __forceinline__ v16h load_frag(const _Float16* __restrict__ blk,
                                          int lane) {
  return *(const v16h*)(blk + lane * 16);   // 32B -> 2x ds_load_b128
}

// A-matrix (16x32 f16) fragment slot for element (row m, k-within-tile kk):
// lanes 0-15: M=lane, elems {0..7}=K0..7, {8..15}=K16..23 ; lanes16-31: K+8.
__device__ __forceinline__ int a_off(int m, int kk) {
  int lane = (m & 15) + (((kk >> 3) & 1) << 4);
  int elem = (kk & 7) + (((kk >> 4) & 1) << 3);
  return lane * 16 + elem;
}

// B-matrix (32x16 f16) fragment slot for element (col n, k-within-tile kk):
// lanes 0-15: N=lane, elems {0..7}=K0..7, {8..15}=K16..23 ; lanes16-31: K+8.
__device__ __forceinline__ int b_off(int n, int kk) {
  int lane = (n & 15) + (((kk >> 3) & 1) << 4);
  int elem = (kk & 7) + (((kk >> 4) & 1) << 3);
  return lane * 16 + elem;
}

// ---------------------------------------------------------------------------
// init: convert x input to f16 workspace, zero aggregation buffers
// ---------------------------------------------------------------------------
__global__ void init_kernel(const float* __restrict__ xin,
                            _Float16* __restrict__ x,
                            float* __restrict__ aggp, float* __restrict__ aggf,
                            int n) {
  int i = blockIdx.x * blockDim.x + threadIdx.x;
  if (i < n) { x[i] = (_Float16)xin[i]; aggp[i] = 0.f; aggf[i] = 0.f; }
}

// ---------------------------------------------------------------------------
// edge encoder (WMMA): e = relu(attr@W1[6,32]+b1)@W2[32,64]+b2 ; past mask
// K=6 is zero-padded to one 32-wide k-tile.
// ---------------------------------------------------------------------------
__global__ __launch_bounds__(128) void encode_kernel(
    const float* __restrict__ ea, const int* __restrict__ src,
    const int* __restrict__ dst, const int* __restrict__ ts,
    const float* __restrict__ w1, const float* __restrict__ b1,
    const float* __restrict__ w2, const float* __restrict__ b2,
    _Float16* __restrict__ e, float* __restrict__ pastf, int nEdges) {
  __shared__ __align__(32) _Float16 sW1f[2 * FRAG];   // 1kt x 2nt (K pad 6->32)
  __shared__ __align__(32) _Float16 sW2f[4 * FRAG];   // 1kt x 4nt (K=32,N=64)
  __shared__ float sB1[32], sB2[64];
  __shared__ __align__(32) _Float16 stA[WAVES_PER_BLOCK][FRAG];
  __shared__ __align__(32) _Float16 stH[WAVES_PER_BLOCK][FRAG];
  const int tid = threadIdx.x;
  const int lane = tid & 31;
  const int wv = tid >> 5;
  const int tileBase = (blockIdx.x * WAVES_PER_BLOCK + wv) * 16;

  // phase 1: zero-fill padded fragments
  {
    v16h z = {};
    if (tid < 64) *(v16h*)(sW1f + tid * 16) = z;      // 2*FRAG = 64 lanes' worth
    *(v16h*)(stA[wv] + lane * 16) = z;
  }
  __syncthreads();

  // phase 2: scatter weights / activations into fragment slots
  for (int i = tid; i < 6 * 32; i += 128) {           // W1: K=6, N=32
    int k = i >> 5, n = i & 31;
    sW1f[(n >> 4) * FRAG + b_off(n, k)] = (_Float16)w1[i];
  }
  for (int i = tid; i < 32 * 64; i += 128) {          // W2: K=32, N=64
    int k = i >> 6, n = i & 63;
    sW2f[(n >> 4) * FRAG + b_off(n, k)] = (_Float16)w2[i];
  }
  if (tid < 32) sB1[tid] = b1[tid];
  if (tid < 64) sB2[tid] = b2[tid];
#pragma unroll
  for (int it = 0; it < 3; ++it) {                    // 16 rows x 6 attrs
    int idx = it * 32 + lane;
    int r = idx / 6, c = idx - r * 6;
    if (r < 16) {
      int eid = tileBase + r;
      float v = (eid < nEdges) ? ea[(size_t)eid * 6 + c] : 0.f;
      stA[wv][a_off(r, c)] = (_Float16)v;
    }
  }
  if (lane < 16) {
    int eid = tileBase + lane;
    if (eid < nEdges)
      pastf[eid] = (ts[src[eid]] <= ts[dst[eid]]) ? 1.f : 0.f;
  }
  __syncthreads();

  // layer 1: [16,32(pad)] @ [32,32]
  v8f acc1[2] = {};
  {
    v16h a = load_frag(stA[wv], lane);
#pragma unroll
    for (int nt = 0; nt < 2; ++nt) {
      v16h b = load_frag(sW1f + nt * FRAG, lane);
      acc1[nt] = wmma_f32_f16(a, b, acc1[nt]);
    }
  }
#pragma unroll
  for (int nt = 0; nt < 2; ++nt) {
    int col = (lane & 15) + nt * 16;
    float bias = sB1[col];
#pragma unroll
    for (int v = 0; v < 8; ++v) {
      int row = v + ((lane >> 4) << 3);
      stH[wv][a_off(row, col)] = (_Float16)fmaxf(acc1[nt][v] + bias, 0.f);
    }
  }
  __syncthreads();

  // layer 2: [16,32] @ [32,64] -> e (no relu on encoder output)
  v8f acc2[4] = {};
  {
    v16h a = load_frag(stH[wv], lane);
#pragma unroll
    for (int nt = 0; nt < 4; ++nt) {
      v16h b = load_frag(sW2f + nt * FRAG, lane);
      acc2[nt] = wmma_f32_f16(a, b, acc2[nt]);
    }
  }
#pragma unroll
  for (int nt = 0; nt < 4; ++nt) {
    int col = (lane & 15) + nt * 16;
    float bias = sB2[col];
#pragma unroll
    for (int v = 0; v < 8; ++v) {
      int row = v + ((lane >> 4) << 3);
      int eid = tileBase + row;
      if (eid < nEdges) e[(size_t)eid * 64 + col] = (_Float16)(acc2[nt][v] + bias);
    }
  }
}

// ---------------------------------------------------------------------------
// one message-passing step over edges:
//   e   = relu( relu([x_s|x_d|e] @ W1 + b1) @ W2 + b2 )          (WMMA)
//   m   = relu( [x_s|e] @ Wm + bm )                              (WMMA)
//   agg_{past|fut}[dst] += m                                     (f32 atomics)
// ---------------------------------------------------------------------------
__global__ __launch_bounds__(128) void edge_step_kernel(
    const _Float16* __restrict__ x, _Float16* __restrict__ e,
    const int* __restrict__ src, const int* __restrict__ dst,
    const float* __restrict__ pastf,
    const float* __restrict__ w1, const float* __restrict__ b1,
    const float* __restrict__ w2, const float* __restrict__ b2,
    const float* __restrict__ wm, const float* __restrict__ bm,
    float* __restrict__ agg_past, float* __restrict__ agg_fut, int nEdges) {
  __shared__ __align__(32) _Float16 sW1f[16 * FRAG];  // 4kt x 4nt, 16 KB
  __shared__ __align__(32) _Float16 sW2f[8 * FRAG];   // 2kt x 4nt,  8 KB
  __shared__ __align__(32) _Float16 sWmf[6 * FRAG];   // 3kt x 2nt,  6 KB
  __shared__ float sB1[64], sB2[64], sBm[32];
  __shared__ __align__(32) _Float16 stA[WAVES_PER_BLOCK][4 * FRAG]; // 16 KB
  __shared__ __align__(32) _Float16 stH[WAVES_PER_BLOCK][2 * FRAG]; //  8 KB
  __shared__ int   sDst[WAVES_PER_BLOCK][16];
  __shared__ float sPst[WAVES_PER_BLOCK][16];

  const int tid = threadIdx.x;
  const int lane = tid & 31;
  const int wv = tid >> 5;

  // stage + swizzle weights (one-time per block)
  for (int i = tid; i < 128 * 64; i += 128) {
    int k = i >> 6, n = i & 63;
    sW1f[((k >> 5) * 4 + (n >> 4)) * FRAG + b_off(n, k & 31)] = (_Float16)w1[i];
  }
  for (int i = tid; i < 64 * 64; i += 128) {
    int k = i >> 6, n = i & 63;
    sW2f[((k >> 5) * 4 + (n >> 4)) * FRAG + b_off(n, k & 31)] = (_Float16)w2[i];
  }
  for (int i = tid; i < 96 * 32; i += 128) {
    int k = i >> 5, n = i & 31;
    sWmf[((k >> 5) * 2 + (n >> 4)) * FRAG + b_off(n, k & 31)] = (_Float16)wm[i];
  }
  if (tid < 64) { sB1[tid] = b1[tid]; sB2[tid] = b2[tid]; }
  if (tid < 32) sBm[tid] = bm[tid];

  const int tileBase = (blockIdx.x * WAVES_PER_BLOCK + wv) * 16;
  _Float16* myA = stA[wv];
  _Float16* myH = stH[wv];

  // stage A = [x_src | x_dst | e0 | e1] -> fragment blocks 0..3 (kk == lane)
  // activations are already f16 in HBM: straight copy, no conversion
#pragma unroll
  for (int r = 0; r < 16; ++r) {
    int eid = tileBase + r;
    int es = (eid < nEdges) ? eid : 0;
    int si = src[es], di = dst[es];
    if (lane == 0) { sDst[wv][r] = di; sPst[wv][r] = pastf[es]; }
    _Float16 vs = x[si * 32 + lane];
    _Float16 vd = x[di * 32 + lane];
    _Float16 v0 = e[(size_t)es * 64 + lane];
    _Float16 v1 = e[(size_t)es * 64 + 32 + lane];
    int o = a_off(r, lane);
    myA[0 * FRAG + o] = vs;
    myA[1 * FRAG + o] = vd;
    myA[2 * FRAG + o] = v0;
    myA[3 * FRAG + o] = v1;
  }
  __syncthreads();

  // ---- layer 1: [16,128] @ [128,64] ----
  v8f acc[4] = {};
#pragma unroll
  for (int kt = 0; kt < 4; ++kt) {
    v16h a = load_frag(myA + kt * FRAG, lane);
#pragma unroll
    for (int nt = 0; nt < 4; ++nt) {
      v16h b = load_frag(sW1f + (kt * 4 + nt) * FRAG, lane);
      acc[nt] = wmma_f32_f16(a, b, acc[nt]);
    }
  }
#pragma unroll
  for (int nt = 0; nt < 4; ++nt) {
    int col = (lane & 15) + nt * 16;
    float bias = sB1[col];
#pragma unroll
    for (int v = 0; v < 8; ++v) {
      int row = v + ((lane >> 4) << 3);
      myH[(col >> 5) * FRAG + a_off(row, col & 31)] =
          (_Float16)fmaxf(acc[nt][v] + bias, 0.f);
    }
  }
  __syncthreads();

  // ---- layer 2: [16,64] @ [64,64] -> new e ----
  v8f acc2[4] = {};
#pragma unroll
  for (int kt = 0; kt < 2; ++kt) {
    v16h a = load_frag(myH + kt * FRAG, lane);
#pragma unroll
    for (int nt = 0; nt < 4; ++nt) {
      v16h b = load_frag(sW2f + (kt * 4 + nt) * FRAG, lane);
      acc2[nt] = wmma_f32_f16(a, b, acc2[nt]);
    }
  }
#pragma unroll
  for (int nt = 0; nt < 4; ++nt) {
    int col = (lane & 15) + nt * 16;
    float bias = sB2[col];
#pragma unroll
    for (int v = 0; v < 8; ++v) {
      int row = v + ((lane >> 4) << 3);
      _Float16 hh = (_Float16)fmaxf(acc2[nt][v] + bias, 0.f);
      int eid = tileBase + row;
      if (eid < nEdges) e[(size_t)eid * 64 + col] = hh;  // in-place e update
      int f = 32 + col;  // msg input feature; block 0 (x_src) stays valid
      myA[(f >> 5) * FRAG + a_off(row, f & 31)] = hh;
    }
  }
  __syncthreads();

  // ---- message: [16,96] @ [96,32] (A = blocks 0..2), scatter f32 atomics ----
  v8f accm[2] = {};
#pragma unroll
  for (int kt = 0; kt < 3; ++kt) {
    v16h a = load_frag(myA + kt * FRAG, lane);
#pragma unroll
    for (int nt = 0; nt < 2; ++nt) {
      v16h b = load_frag(sWmf + (kt * 2 + nt) * FRAG, lane);
      accm[nt] = wmma_f32_f16(a, b, accm[nt]);
    }
  }
#pragma unroll
  for (int nt = 0; nt < 2; ++nt) {
    int col = (lane & 15) + nt * 16;
    float bias = sBm[col];
#pragma unroll
    for (int v = 0; v < 8; ++v) {
      int row = v + ((lane >> 4) << 3);
      int eid = tileBase + row;
      if (eid < nEdges) {
        float m = fmaxf(accm[nt][v] + bias, 0.f);
        // past/fut masks are complementary 0/1: one atomic per element
        float* tgt = (sPst[wv][row] > 0.5f) ? agg_past : agg_fut;
        __hip_atomic_fetch_add(&tgt[(size_t)sDst[wv][row] * 32 + col], m,
                               __ATOMIC_RELAXED, __HIP_MEMORY_SCOPE_AGENT);
      }
    }
  }
}

// ---------------------------------------------------------------------------
// node update: x = relu([agg_p|agg_f|init_x] @ Wn + bn) ; re-zero agg buffers
// ---------------------------------------------------------------------------
__global__ __launch_bounds__(128) void node_step_kernel(
    float* __restrict__ agg_past, float* __restrict__ agg_fut,
    const float* __restrict__ initx,
    const float* __restrict__ wn, const float* __restrict__ bn,
    _Float16* __restrict__ x, int nNodes) {
  __shared__ __align__(32) _Float16 sWnf[6 * FRAG];               // 3kt x 2nt
  __shared__ float sBn[32];
  __shared__ __align__(32) _Float16 stA[WAVES_PER_BLOCK][3 * FRAG];
  const int tid = threadIdx.x;
  const int lane = tid & 31;
  const int wv = tid >> 5;
  for (int i = tid; i < 96 * 32; i += 128) {
    int k = i >> 5, n = i & 31;
    sWnf[((k >> 5) * 2 + (n >> 4)) * FRAG + b_off(n, k & 31)] = (_Float16)wn[i];
  }
  if (tid < 32) sBn[tid] = bn[tid];
  const int tileBase = (blockIdx.x * WAVES_PER_BLOCK + wv) * 16;
  _Float16* myA = stA[wv];
#pragma unroll
  for (int r = 0; r < 16; ++r) {
    int nid = tileBase + r;
    int ns = (nid < nNodes) ? nid : 0;
    float ap = agg_past[ns * 32 + lane];
    float af = agg_fut[ns * 32 + lane];
    float ix = initx[ns * 32 + lane];
    if (nid < nNodes) {  // reset accumulators for next step
      agg_past[nid * 32 + lane] = 0.f;
      agg_fut[nid * 32 + lane] = 0.f;
    }
    int o = a_off(r, lane);
    myA[0 * FRAG + o] = (_Float16)ap;
    myA[1 * FRAG + o] = (_Float16)af;
    myA[2 * FRAG + o] = (_Float16)ix;
  }
  __syncthreads();
  v8f acc[2] = {};
#pragma unroll
  for (int kt = 0; kt < 3; ++kt) {
    v16h a = load_frag(myA + kt * FRAG, lane);
#pragma unroll
    for (int nt = 0; nt < 2; ++nt) {
      v16h b = load_frag(sWnf + (kt * 2 + nt) * FRAG, lane);
      acc[nt] = wmma_f32_f16(a, b, acc[nt]);
    }
  }
#pragma unroll
  for (int nt = 0; nt < 2; ++nt) {
    int col = (lane & 15) + nt * 16;
    float bias = sBn[col];
#pragma unroll
    for (int v = 0; v < 8; ++v) {
      int row = v + ((lane >> 4) << 3);
      int nid = tileBase + row;
      if (nid < nNodes)
        x[nid * 32 + col] = (_Float16)fmaxf(acc[nt][v] + bias, 0.f);
    }
  }
}

// ---------------------------------------------------------------------------
// classifier (WMMA): sigmoid( relu(e@W1[64,64]+b1) @ W2[64,1] + b2 )
// layer 2 = per-lane partial dot over 4 columns + 16-wide LDS reduction.
// ---------------------------------------------------------------------------
__global__ __launch_bounds__(128) void cls_kernel(
    const _Float16* __restrict__ e, const float* __restrict__ w1,
    const float* __restrict__ b1, const float* __restrict__ w2,
    const float* __restrict__ b2, float* __restrict__ out, int nEdges) {
  __shared__ __align__(32) _Float16 sW1f[8 * FRAG];   // 2kt x 4nt
  __shared__ float sB1[64], sW2[64], sB2s[1];
  __shared__ __align__(32) _Float16 stA[WAVES_PER_BLOCK][2 * FRAG];
  __shared__ float sPart[WAVES_PER_BLOCK][16][16];
  const int tid = threadIdx.x;
  const int lane = tid & 31;
  const int wv = tid >> 5;
  for (int i = tid; i < 64 * 64; i += 128) {
    int k = i >> 6, n = i & 63;
    sW1f[((k >> 5) * 4 + (n >> 4)) * FRAG + b_off(n, k & 31)] = (_Float16)w1[i];
  }
  if (tid < 64) { sB1[tid] = b1[tid]; sW2[tid] = w2[tid]; }
  if (tid == 0) sB2s[0] = b2[0];
  const int tileBase = (blockIdx.x * WAVES_PER_BLOCK + wv) * 16;
#pragma unroll
  for (int r = 0; r < 16; ++r) {
    int eid = tileBase + r;
    int es = (eid < nEdges) ? eid : 0;
    _Float16 v0 = e[(size_t)es * 64 + lane];
    _Float16 v1 = e[(size_t)es * 64 + 32 + lane];
    int o = a_off(r, lane);
    stA[wv][0 * FRAG + o] = v0;
    stA[wv][1 * FRAG + o] = v1;
  }
  __syncthreads();
  v8f acc[4] = {};
#pragma unroll
  for (int kt = 0; kt < 2; ++kt) {
    v16h a = load_frag(stA[wv] + kt * FRAG, lane);
#pragma unroll
    for (int nt = 0; nt < 4; ++nt) {
      v16h b = load_frag(sW1f + (kt * 4 + nt) * FRAG, lane);
      acc[nt] = wmma_f32_f16(a, b, acc[nt]);
    }
  }
  // partial dot with w2 over this lane's 4 columns
#pragma unroll
  for (int v = 0; v < 8; ++v) {
    int row = v + ((lane >> 4) << 3);
    float p = 0.f;
#pragma unroll
    for (int nt = 0; nt < 4; ++nt) {
      int col = (lane & 15) + nt * 16;
      p += fmaxf(acc[nt][v] + sB1[col], 0.f) * sW2[col];
    }
    sPart[wv][row][lane & 15] = p;
  }
  __syncthreads();
  if (lane < 16) {
    int eid = tileBase + lane;
    if (eid < nEdges) {
      float z = sB2s[0];
#pragma unroll
      for (int c = 0; c < 16; ++c) z += sPart[wv][lane][c];
      out[eid] = 1.f / (1.f + __expf(-z));
    }
  }
}

// ---------------------------------------------------------------------------
extern "C" void kernel_launch(void* const* d_in, const int* in_sizes, int n_in,
                              void* d_out, int out_size, void* d_ws,
                              size_t ws_size, hipStream_t stream) {
  const float* x_in      = (const float*)d_in[0];
  const int*   edge_idx  = (const int*)d_in[1];
  const float* edge_attr = (const float*)d_in[2];
  const int*   node_ts   = (const int*)d_in[3];
  const float* initial_x = (const float*)d_in[4];
  const float* enc_w1 = (const float*)d_in[5];
  const float* enc_b1 = (const float*)d_in[6];
  const float* enc_w2 = (const float*)d_in[7];
  const float* enc_b2 = (const float*)d_in[8];
  const float* edge_w1 = (const float*)d_in[9];
  const float* edge_b1 = (const float*)d_in[10];
  const float* edge_w2 = (const float*)d_in[11];
  const float* edge_b2 = (const float*)d_in[12];
  const float* msg_w = (const float*)d_in[13];
  const float* msg_b = (const float*)d_in[14];
  const float* node_w = (const float*)d_in[15];
  const float* node_b = (const float*)d_in[16];
  const float* cls_w1 = (const float*)d_in[17];
  const float* cls_b1 = (const float*)d_in[18];
  const float* cls_w2 = (const float*)d_in[19];
  const float* cls_b2 = (const float*)d_in[20];

  const int N = in_sizes[0] / 32;
  const int E = in_sizes[2] / 6;
  const int* src = edge_idx;
  const int* dst = edge_idx + E;

  // workspace layout (256B-aligned regions):
  //   e[E,64] f16 | x[N,32] f16 | agg_p[N,32] f32 | agg_f[N,32] f32 | past[E] f32
  char* ws = (char*)d_ws;
  size_t off = 0;
  auto align256 = [](size_t v) { return (v + 255) & ~(size_t)255; };
  _Float16* e_buf = (_Float16*)(ws + off);
  off = align256(off + (size_t)E * 64 * sizeof(_Float16));
  _Float16* x_buf = (_Float16*)(ws + off);
  off = align256(off + (size_t)N * 32 * sizeof(_Float16));
  float* aggp = (float*)(ws + off);
  off = align256(off + (size_t)N * 32 * sizeof(float));
  float* aggf = (float*)(ws + off);
  off = align256(off + (size_t)N * 32 * sizeof(float));
  float* pastf = (float*)(ws + off);

  const int tileRows = 16 * WAVES_PER_BLOCK;     // 64 rows per block
  const int edgeBlocks = (E + tileRows - 1) / tileRows;
  const int nodeBlocks = (N + tileRows - 1) / tileRows;

  init_kernel<<<(N * 32 + 255) / 256, 256, 0, stream>>>(x_in, x_buf, aggp, aggf,
                                                        N * 32);
  encode_kernel<<<edgeBlocks, 128, 0, stream>>>(
      edge_attr, src, dst, node_ts, enc_w1, enc_b1, enc_w2, enc_b2, e_buf,
      pastf, E);
  for (int step = 0; step < 12; ++step) {
    edge_step_kernel<<<edgeBlocks, 128, 0, stream>>>(
        x_buf, e_buf, src, dst, pastf, edge_w1, edge_b1, edge_w2, edge_b2,
        msg_w, msg_b, aggp, aggf, E);
    node_step_kernel<<<nodeBlocks, 128, 0, stream>>>(aggp, aggf, initial_x,
                                                     node_w, node_b, x_buf, N);
  }
  cls_kernel<<<edgeBlocks, 128, 0, stream>>>(e_buf, cls_w1, cls_b1, cls_w2,
                                             cls_b2, (float*)d_out, E);
}
